// GlobalAttention_19464791785879
// MI455X (gfx1250) — matrix operand order repaired and checked by reading, not compiled, not run on hardware
//
#include <hip/hip_runtime.h>
#include <hip/hip_bf16.h>
#include <math.h>

#define Bd 16
#define Ld 4096
#define Hd 1024
#define MTILE 128
#define LDA 1032   // padded bf16 K-stride: row stride 2064B -> lanes map to distinct LDS bank groups

typedef __attribute__((ext_vector_type(16))) __bf16 v16bf;
typedef __attribute__((ext_vector_type(8)))  __bf16 v8bf;
typedef __attribute__((ext_vector_type(8)))  float  v8f;

// ---------------------------------------------------------------------------
// Prep 1: weT[n][k] = bf16( attn_w[Hd + k][n] )   (w_e transposed, 2 MB -> L2)
// ---------------------------------------------------------------------------
__global__ __launch_bounds__(256)
void k_weT(const float* __restrict__ attn_w, __bf16* __restrict__ weT) {
  int idx = blockIdx.x * 256 + threadIdx.x;     // 1M elements, grid 4096
  int n = idx >> 10;
  int k = idx & 1023;
  weT[idx] = (__bf16)attn_w[(size_t)(Hd + k) * Hd + n];
}

// ---------------------------------------------------------------------------
// Prep 2: proj_h[b][k] = sum_h rep_h[b][h] * attn_w[h][k] + attn_b[k]
// rep_h[b][h] = hidden[b][h % 512]
// ---------------------------------------------------------------------------
__global__ __launch_bounds__(256)
void k_projh(const float* __restrict__ hidden, const float* __restrict__ attn_w,
             const float* __restrict__ attn_b, float* __restrict__ projh) {
  const int b = blockIdx.y;
  const int k = blockIdx.x * 256 + threadIdx.x;
  __shared__ float sh[512];
  for (int i = threadIdx.x; i < 512; i += 256) sh[i] = hidden[b * 512 + i];
  __syncthreads();
  float acc = attn_b[k];
  for (int h = 0; h < Hd; ++h)
    acc += sh[h & 511] * attn_w[(size_t)h * Hd + k];
  projh[b * Hd + k] = acc;
}

// ---------------------------------------------------------------------------
// Main fused kernel: scores[b][l] = v_w . tanh(proj_h[b] + enc[b,l,:] @ w_e)
// One workgroup = one batch x 128 rows of L. enc tile staged in LDS as bf16.
// 8 waves x 8 n-blocks each; per n-block 8 M-tiles accumulate via
// v_wmma_f32_16x16x32_bf16. A fragments are double-buffered across the
// M-tile loop and B fragments across the K loop so no load ever writes the
// VGPRs a still-in-flight WMMA is reading (kills the 4x v_nop WAR padding)
// and the L2 B-fetch latency is covered by the 8-WMMA inner loop.
// ---------------------------------------------------------------------------
extern __shared__ char smem_raw[];

__device__ __forceinline__ v16bf loadA(const __bf16* sA, int mt, int l15,
                                       int kb, int half) {
  const __bf16* ap = sA + (size_t)(mt * 16 + l15) * LDA + kb * 32 + half * 8;
  v8bf alo = *(const v8bf*)ap;
  v8bf ahi = *(const v8bf*)(ap + 16);
  v16bf a;
#pragma unroll
  for (int i = 0; i < 8; ++i) { a[i] = alo[i]; a[8 + i] = ahi[i]; }
  return a;
}

__global__ __launch_bounds__(256)
void k_scores(const float* __restrict__ enc, const __bf16* __restrict__ weT,
              const float* __restrict__ projh, const float* __restrict__ vw,
              float* __restrict__ scores) {
  __bf16* sA     = (__bf16*)smem_raw;                                  // [128][1032] bf16
  float*  sScore = (float*)(smem_raw + (size_t)MTILE * LDA * 2);       // [128]

  const int b    = blockIdx.y;
  const int l0   = blockIdx.x * MTILE;
  const int tid  = threadIdx.x;
  const int lane = tid & 31;
  const int wave = tid >> 5;
  const int half = lane >> 4;     // K-half selector (ISA 16-bit A/B layouts)
  const int l15  = lane & 15;     // A: row within tile / B,C: column N

  if (tid < MTILE) sScore[tid] = 0.0f;

  // ---- stage 128x1024 f32 tile -> bf16 LDS (coalesced float4 loads) ----
  const float4* src = (const float4*)(enc + ((size_t)b * Ld + l0) * Hd);
  for (int i = tid; i < MTILE * (Hd / 4); i += 256) {
    int row = i >> 8;          // Hd/4 = 256 float4 per row
    int c4  = i & 255;
    float4 v = src[(size_t)row * 256 + c4];
    __bf16* dst = sA + (size_t)row * LDA + c4 * 4;
    dst[0] = (__bf16)v.x; dst[1] = (__bf16)v.y;
    dst[2] = (__bf16)v.z; dst[3] = (__bf16)v.w;
  }
  __syncthreads();

  for (int nbi = 0; nbi < 8; ++nbi) {
    const int nb = wave * 8 + nbi;     // 64 n-blocks of 16 columns
    const int n0 = nb * 16;

    v8f acc[8];
#pragma unroll
    for (int mt = 0; mt < 8; ++mt)
#pragma unroll
      for (int i = 0; i < 8; ++i) acc[mt][i] = 0.0f;

    // B fragment base: column n = n0+l15, K = kb*32 + half*8 + {0..7} and kb*32 + 16 + half*8 + {0..7}
    const __bf16* bptr = weT + (size_t)(n0 + l15) * Hd + half * 8;

    v16bf bcur = __builtin_shufflevector(*(const v8bf*)(bptr), *(const v8bf*)(bptr + 16), 0, 1, 2, 3, 4, 5, 6, 7, 8, 9, 10, 11, 12, 13, 14, 15);          // kb = 0
    for (int kb = 0; kb < 32; ++kb) {
      // prefetch next K-step's B fragment; full mt loop hides the L2 latency
      const int kbn = (kb < 31) ? kb + 1 : kb;
      v16bf bnext = __builtin_shufflevector(*(const v8bf*)(bptr + kbn * 32), *(const v8bf*)(bptr + kbn * 32 + 16), 0, 1, 2, 3, 4, 5, 6, 7, 8, 9, 10, 11, 12, 13, 14, 15);

      v16bf acur = loadA(sA, 0, l15, kb, half);
#pragma unroll
      for (int mt = 0; mt < 8; ++mt) {
        // load mt+1's A into a fresh buffer BEFORE consuming acur:
        // avoids the WMMA->VALU WAR hazard (no v_nop padding)
        const int mtn = (mt < 7) ? mt + 1 : mt;
        v16bf anext = loadA(sA, mtn, l15, kb, half);
        acc[mt] = __builtin_amdgcn_wmma_f32_16x16x32_bf16(
            false, acur, false, bcur, (short)0, acc[mt], false, false);
        acur = anext;
      }
      bcur = bnext;
    }

    // ---- epilogue: tanh + dot with v_w, reduce over the 16 N-lanes ----
    const float ph  = projh[b * Hd + n0 + l15];
    const float vwv = vw[n0 + l15];
#pragma unroll
    for (int mt = 0; mt < 8; ++mt) {
#pragma unroll
      for (int r = 0; r < 8; ++r) {
        // C layout: lane l15 = column N, VGPR r = row (half*8 + r)
        float e = tanhf(acc[mt][r] + ph) * vwv;
        e += __shfl_xor(e, 1, 32);
        e += __shfl_xor(e, 2, 32);
        e += __shfl_xor(e, 4, 32);
        e += __shfl_xor(e, 8, 32);
        if (l15 == 0)
          atomicAdd(&sScore[mt * 16 + half * 8 + r], e);
      }
    }
  }
  __syncthreads();
  if (tid < MTILE) scores[(size_t)b * Ld + l0 + tid] = sScore[tid];
}

// ---------------------------------------------------------------------------
// Softmax over L, in place. One block per batch.
// ---------------------------------------------------------------------------
__global__ __launch_bounds__(256)
void k_softmax(float* __restrict__ scores) {
  const int b = blockIdx.x;
  float* s = scores + (size_t)b * Ld;
  __shared__ float red[256];
  const int tid = threadIdx.x;

  float m = -INFINITY;
  for (int l = tid; l < Ld; l += 256) m = fmaxf(m, s[l]);
  red[tid] = m; __syncthreads();
  for (int off = 128; off > 0; off >>= 1) {
    if (tid < off) red[tid] = fmaxf(red[tid], red[tid + off]);
    __syncthreads();
  }
  m = red[0]; __syncthreads();

  float sum = 0.0f;
  for (int l = tid; l < Ld; l += 256) { float e = __expf(s[l] - m); s[l] = e; sum += e; }
  red[tid] = sum; __syncthreads();
  for (int off = 128; off > 0; off >>= 1) {
    if (tid < off) red[tid] += red[tid + off];
    __syncthreads();
  }
  const float inv = 1.0f / red[0];
  for (int l = tid; l < Ld; l += 256) s[l] *= inv;
}

// ---------------------------------------------------------------------------
// context[b][h] = sum_l weights[b][l] * enc[b][l][h]   (bandwidth bound)
// ---------------------------------------------------------------------------
__global__ __launch_bounds__(256)
void k_zero(float* __restrict__ out, int n) {
  int i = blockIdx.x * 256 + threadIdx.x;
  if (i < n) out[i] = 0.0f;
}

__global__ __launch_bounds__(256)
void k_context(const float* __restrict__ enc, const float* __restrict__ w,
               float* __restrict__ out) {
  const int b  = blockIdx.z;
  const int h  = blockIdx.x * 256 + threadIdx.x;
  const int l0 = blockIdx.y * 512;
  const float* wb = w + (size_t)b * Ld + l0;
  const float* e  = enc + ((size_t)b * Ld + l0) * Hd + h;
  float acc = 0.0f;
  for (int l = 0; l < 512; ++l)
    acc += wb[l] * e[(size_t)l * Hd];
  atomicAdd(&out[b * Hd + h], acc);
}

// ---------------------------------------------------------------------------
extern "C" void kernel_launch(void* const* d_in, const int* in_sizes, int n_in,
                              void* d_out, int out_size, void* d_ws, size_t ws_size,
                              hipStream_t stream) {
  const float* hidden = (const float*)d_in[0];   // (16, 512)
  const float* enc    = (const float*)d_in[1];   // (16, 4096, 1024)
  const float* attn_w = (const float*)d_in[2];   // (2048, 1024)
  const float* attn_b = (const float*)d_in[3];   // (1024,)
  const float* v_w    = (const float*)d_in[4];   // (1024,)
  float* out = (float*)d_out;                    // (16, 1024)

  char* ws = (char*)d_ws;
  __bf16* weT   = (__bf16*)ws;                                 // 2 MB
  float*  projh = (float*)(ws + (2u << 20));                   // 64 KB
  float*  scores = (float*)(ws + (2u << 20) + (64u << 10));    // 256 KB

  k_weT  <<<4096, 256, 0, stream>>>(attn_w, weT);
  k_projh<<<dim3(Hd / 256, Bd), 256, 0, stream>>>(hidden, attn_w, attn_b, projh);

  const size_t smem = (size_t)MTILE * LDA * 2 + MTILE * 4;     // ~258.5 KB (<=320 KB/WGP)
  k_scores<<<dim3(Ld / MTILE, Bd), 256, smem, stream>>>(enc, weT, projh, v_w, scores);

  k_softmax<<<Bd, 256, 0, stream>>>(scores);

  k_zero   <<<(Bd * Hd + 255) / 256, 256, 0, stream>>>(out, Bd * Hd);
  k_context<<<dim3(Hd / 256, Ld / 512, Bd), 256, 0, stream>>>(enc, scores, out);
}
